// MyModel_22814866277033
// MI455X (gfx1250) — compile-verified, matrix operand order checked by reading, not run on hardware
//
#include <hip/hip_runtime.h>

// ---------------------------------------------------------------------------
// MI455X (gfx1250, wave32) implementation of the CNN + Product-Key-Memory net.
// Large GEMMs use v_wmma_f32_16x16x32_f16; dense1 stages its shared B panel
// into LDS with the Tensor Data Mover (tensor_load_to_lds + s_wait_tensorcnt).
// ---------------------------------------------------------------------------

#define BATCH   4096
#define HEADS   4
#define KNN     32
#define NKEYS   512
#define VDIM    128
#define BN_EPS  1e-5f

typedef _Float16 h16;
typedef __attribute__((ext_vector_type(16))) _Float16 v16h;
typedef __attribute__((ext_vector_type(8)))  _Float16 v8h;
typedef __attribute__((ext_vector_type(8)))  float    v8f;
typedef __attribute__((ext_vector_type(4)))  unsigned int u32x4;
typedef __attribute__((ext_vector_type(8)))  int          i32x8;
typedef __attribute__((ext_vector_type(4)))  int          i32x4;

// Build a 16-half WMMA fragment from two contiguous 8-half (16B) runs.
// Per CDNA5 ISA 7.12.2, lane L of an A/B fragment holds K = kb..kb+7 in
// vgprs 0..3 and K = kb+16..kb+23 in vgprs 4..7 (kb = (L>>4)*8); the ext
// vector's elements 0..7 / 8..15 map to exactly those register pairs.
__device__ __forceinline__ v16h frag_load(const h16* __restrict__ p0,
                                          const h16* __restrict__ p1) {
  v8h a = *(const v8h*)p0;
  v8h b = *(const v8h*)p1;
  v16h r;
#pragma unroll
  for (int i = 0; i < 8; ++i) { r[i] = a[i]; r[i + 8] = b[i]; }
  return r;
}

__device__ __forceinline__ v8f wmma_f16(v16h a, v16h b, v8f c) {
  return __builtin_amdgcn_wmma_f32_16x16x32_f16(
      /*neg_a=*/false, a, /*neg_b=*/false, b,
      /*c_mod=*/(short)0, c, /*reuse_a=*/false, /*reuse_b=*/false);
}

// ---------------------------------------------------------------------------
// conv1 (3x3, 1->32, VALID) + ReLU, fp32 in -> f16 out  [B,26,26,32]
// ---------------------------------------------------------------------------
__global__ void k_conv1(const float* __restrict__ x, const float* __restrict__ w,
                        const float* __restrict__ bias, h16* __restrict__ h1) {
  int idx = blockIdx.x * blockDim.x + threadIdx.x;
  const int total = BATCH * 26 * 26 * 32;
  if (idx >= total) return;
  int c  = idx & 31;
  int p  = idx >> 5;
  int xx = p % 26; int t = p / 26;
  int yy = t % 26; int b = t / 26;
  float acc = bias[c];
  const float* xb = x + ((size_t)b * 28 + yy) * 28 + xx;
#pragma unroll
  for (int ky = 0; ky < 3; ++ky)
#pragma unroll
    for (int kx = 0; kx < 3; ++kx)
      acc += xb[ky * 28 + kx] * w[(ky * 3 + kx) * 32 + c];
  h1[idx] = (h16)(acc > 0.f ? acc : 0.f);
}

// ---------------------------------------------------------------------------
// Weight prep: src fp32 [K][N] row-major -> dst f16 [N][K] (K contiguous),
// which is exactly the B-fragment-friendly layout.
// ---------------------------------------------------------------------------
__global__ void k_transpose_cast(const float* __restrict__ src,
                                 h16* __restrict__ dst, int K, int N) {
  int idx = blockIdx.x * blockDim.x + threadIdx.x;
  if (idx >= K * N) return;
  int n = idx % N, k = idx / N;
  dst[(size_t)n * K + k] = (h16)src[(size_t)k * N + n];
}

__global__ void k_cast_f16(const float* __restrict__ src, h16* __restrict__ dst,
                           int n) {
  int idx = blockIdx.x * blockDim.x + threadIdx.x;
  if (idx < n) dst[idx] = (h16)src[idx];
}

// ---------------------------------------------------------------------------
// conv2 as implicit GEMM: M = B*24*24 pixels, N = 64 ch, K = 288 (= 9 k-steps).
// im2col row k index = (ky*3+kx)*32 + ci  (matches HWIO weight flattening).
// One wave -> 16 pixels x 64 channels (4 WMMA tiles).  Output f16.
// ---------------------------------------------------------------------------
__global__ __launch_bounds__(256) void k_conv2(const h16* __restrict__ h1,
                                               const h16* __restrict__ w2t,
                                               const float* __restrict__ bias,
                                               h16* __restrict__ c2) {
  const int lane  = threadIdx.x & 31;
  const int wave  = threadIdx.x >> 5;
  const int mtile = blockIdx.x * 8 + wave;          // 147456 tiles of 16 pixels
  const int kb    = (lane >> 4) << 3;               // 0 or 8
  const int m     = lane & 15;

  int pix = mtile * 16 + m;
  int xx  = pix % 24; int t = pix / 24;
  int yy  = t % 24;   int b = t / 24;
  const h16* pA = h1 + (((size_t)b * 26 + yy) * 26 + xx) * 32 + kb;
  const h16* pB = w2t + (size_t)m * 288 + kb;       // col base for n-tile 0

  v8f a0 = {}, a1 = {}, a2 = {}, a3 = {};
#pragma unroll
  for (int kk = 0; kk < 9; ++kk) {
    const h16* ap = pA + ((kk / 3) * 26 + (kk % 3)) * 32;
    v16h af = frag_load(ap, ap + 16);
    const h16* bp = pB + kk * 32;
    a0 = wmma_f16(af, frag_load(bp,            bp + 16),            a0);
    a1 = wmma_f16(af, frag_load(bp + 16 * 288, bp + 16 * 288 + 16), a1);
    a2 = wmma_f16(af, frag_load(bp + 32 * 288, bp + 32 * 288 + 16), a2);
    a3 = wmma_f16(af, frag_load(bp + 48 * 288, bp + 48 * 288 + 16), a3);
  }

  const int row0 = (lane >> 4) * 8;
  const int col  = lane & 15;
#pragma unroll
  for (int v = 0; v < 8; ++v) {
    size_t p = (size_t)mtile * 16 + row0 + v;
    h16* o = c2 + p * 64 + col;
    o[0]  = (h16)(a0[v] + bias[col]);
    o[16] = (h16)(a1[v] + bias[col + 16]);
    o[32] = (h16)(a2[v] + bias[col + 32]);
    o[48] = (h16)(a3[v] + bias[col + 48]);
  }
}

// ---------------------------------------------------------------------------
// 2x2 max-pool [B,24,24,64] -> flattened f16 A-matrix [B][9216]
// ---------------------------------------------------------------------------
__global__ void k_pool(const h16* __restrict__ c2, h16* __restrict__ a1) {
  int idx = blockIdx.x * blockDim.x + threadIdx.x;
  const int total = BATCH * 12 * 12 * 64;
  if (idx >= total) return;
  int c  = idx & 63;
  int p  = idx >> 6;
  int xx = p % 12; int t = p / 12;
  int yy = t % 12; int b = t / 12;
  const h16* s = c2 + (((size_t)b * 24 + yy * 2) * 24 + xx * 2) * 64 + c;
  float m0 = (float)s[0],        m1 = (float)s[64];
  float m2 = (float)s[24 * 64],  m3 = (float)s[24 * 64 + 64];
  float mv = fmaxf(fmaxf(m0, m1), fmaxf(m2, m3));
  a1[(size_t)b * 9216 + (yy * 12 + xx) * 64 + c] = (h16)mv;
}

// ---------------------------------------------------------------------------
// dense1: [4096,9216] x [9216,128] + bias, ReLU.
// The B panel (64 cols x 9216) is shared by all 8 waves in a block, so it is
// staged chunk-by-chunk into LDS with the Tensor Data Mover (double-buffered),
// tracked with TENSORcnt.  A rows stream directly from global.
// ---------------------------------------------------------------------------
#define D1_KC 256                  // halfs per K-chunk (512 B per row)
#define D1_NC (9216 / D1_KC)       // 36 chunks

// Issue one TDM 2-D tile load: 64 rows x 512 B from Bt panel -> LDS buffer.
// This toolchain exposes the 6-arg tensor_load_to_lds
// (g0:u32x4, g1:i32x8, g2:i32x4, g3:i32x4, g4:i32x8, cpol:i32).
__device__ __forceinline__ void tdm_load_panel(const h16* gsrc, unsigned lds_addr) {
  unsigned long long ga = (unsigned long long)(uintptr_t)gsrc;
  u32x4 g0;
  g0[0] = 1u;                                            // count=1, user descriptor
  g0[1] = lds_addr;                                      // LDS byte address
  g0[2] = (unsigned)(ga & 0xFFFFFFFFull);                // global_addr[31:0]
  g0[3] = (unsigned)((ga >> 32) & 0x1FFFFFFull)          // global_addr[56:32]
        | (2u << 30);                                    // type=2 ("image")
  const unsigned td0   = 9216u * 2u / 8u;                // row length, 8B units
  const unsigned td1   = 64u;                            // rows
  const unsigned tile0 = D1_KC * 2u / 8u;                // 64 x 8B = 512 B
  const unsigned tile1 = 64u;                            // 64 rows
  const unsigned str0  = 9216u * 2u / 8u;                // row stride, 8B units
  i32x8 g1;
  g1[0] = (int)(3u << 16);                               // data_size=3 (8 B)
  g1[1] = (int)((td0 & 0xFFFFu) << 16);                  // tensor_dim0[15:0]
  g1[2] = (int)((td0 >> 16) | ((td1 & 0xFFFFu) << 16));  // td0 hi | td1 lo
  g1[3] = (int)((td1 >> 16) | (tile0 << 16));            // td1 hi | tile_dim0
  g1[4] = (int)tile1;                                    // tile_dim1 (tile_dim2=0)
  g1[5] = (int)str0;                                     // dim0 stride [31:0]
  g1[6] = 0;                                             // stride hi | dim1 stride
  g1[7] = 0;
  i32x4 z4 = {0, 0, 0, 0};                               // 2-D: groups 2/3 unused
  i32x8 z8 = {0, 0, 0, 0, 0, 0, 0, 0};
  __builtin_amdgcn_tensor_load_to_lds(g0, g1, z4, z4, z8, 0);
}

__global__ __launch_bounds__(256) void k_dense1(const h16* __restrict__ A,
                                                const h16* __restrict__ Bt,
                                                const float* __restrict__ bias,
                                                float* __restrict__ hf,
                                                h16* __restrict__ hh) {
  __shared__ h16 bsh[2][64][D1_KC];                      // 64 KB double buffer

  const int lane  = threadIdx.x & 31;
  const int wave  = threadIdx.x >> 5;
  const int mtile = blockIdx.x * 8 + wave;   // 256 m-tiles
  const int n0    = blockIdx.y * 64;         // 2 n-groups
  const int kb    = (lane >> 4) << 3;
  const h16* pA    = A  + (size_t)(mtile * 16 + (lane & 15)) * 9216 + kb;
  const h16* panel = Bt + (size_t)n0 * 9216;             // [64][9216] f16
  const unsigned lds0 = (unsigned)(uintptr_t)&bsh[0][0][0];
  const unsigned lds1 = (unsigned)(uintptr_t)&bsh[1][0][0];

  v8f a0 = {}, a1 = {}, a2 = {}, a3 = {};

  if (wave == 0) tdm_load_panel(panel, lds0);            // prime chunk 0

  for (int c = 0; c < D1_NC; ++c) {
    if (wave == 0) {
      if (c + 1 < D1_NC) {
        tdm_load_panel(panel + (c + 1) * D1_KC, ((c + 1) & 1) ? lds1 : lds0);
        __builtin_amdgcn_s_wait_tensorcnt(1);            // chunk c complete
      } else {
        __builtin_amdgcn_s_wait_tensorcnt(0);
      }
    }
    __syncthreads();                                     // chunk c visible to all

    const h16* aa = pA + c * D1_KC;
    const h16* bb = &bsh[c & 1][lane & 15][kb];
#pragma unroll
    for (int k = 0; k < D1_KC; k += 32) {
      v16h af = frag_load(aa + k, aa + k + 16);
      if (k == 0 && c + 1 < D1_NC) __builtin_prefetch(aa + D1_KC, 0, 0);
      a0 = wmma_f16(af, frag_load(bb + k,               bb + k + 16),               a0);
      a1 = wmma_f16(af, frag_load(bb + 16 * D1_KC + k,  bb + 16 * D1_KC + k + 16),  a1);
      a2 = wmma_f16(af, frag_load(bb + 32 * D1_KC + k,  bb + 32 * D1_KC + k + 16),  a2);
      a3 = wmma_f16(af, frag_load(bb + 48 * D1_KC + k,  bb + 48 * D1_KC + k + 16),  a3);
    }
    __syncthreads();                                     // done reading buf[c&1]
  }

  const int row0 = (lane >> 4) * 8;
  const int col  = n0 + (lane & 15);
#pragma unroll
  for (int v = 0; v < 8; ++v) {
    size_t r = (size_t)mtile * 16 + row0 + v;
    float v0 = fmaxf(a0[v] + bias[col],      0.f);
    float v1 = fmaxf(a1[v] + bias[col + 16], 0.f);
    float v2 = fmaxf(a2[v] + bias[col + 32], 0.f);
    float v3 = fmaxf(a3[v] + bias[col + 48], 0.f);
    hf[r * 128 + col]      = v0;  hh[r * 128 + col]      = (h16)v0;
    hf[r * 128 + col + 16] = v1;  hh[r * 128 + col + 16] = (h16)v1;
    hf[r * 128 + col + 32] = v2;  hh[r * 128 + col + 32] = (h16)v2;
    hf[r * 128 + col + 48] = v3;  hh[r * 128 + col + 48] = (h16)v3;
  }
}

// ---------------------------------------------------------------------------
// q = BN(h @ q_w + q_b): [4096,128] x [128,512], batchnorm epilogue -> f16
// ---------------------------------------------------------------------------
__global__ __launch_bounds__(256) void k_q(const h16* __restrict__ A,
                                           const h16* __restrict__ Bt,
                                           const float* __restrict__ qb,
                                           const float* __restrict__ gamma,
                                           const float* __restrict__ beta,
                                           const float* __restrict__ mean,
                                           const float* __restrict__ var,
                                           h16* __restrict__ q) {
  const int lane  = threadIdx.x & 31;
  const int wave  = threadIdx.x >> 5;
  const int mtile = blockIdx.x * 8 + wave;   // 256
  const int n0    = blockIdx.y * 64;         // 8 groups
  const int kb    = (lane >> 4) << 3;
  const h16* pA = A  + (size_t)(mtile * 16 + (lane & 15)) * 128 + kb;
  const h16* pB = Bt + (size_t)(n0 + (lane & 15)) * 128 + kb;

  v8f a0 = {}, a1 = {}, a2 = {}, a3 = {};
#pragma unroll
  for (int k = 0; k < 128; k += 32) {
    v16h af = frag_load(pA + k, pA + k + 16);
    a0 = wmma_f16(af, frag_load(pB + k,            pB + k + 16),            a0);
    a1 = wmma_f16(af, frag_load(pB + 16 * 128 + k, pB + 16 * 128 + k + 16), a1);
    a2 = wmma_f16(af, frag_load(pB + 32 * 128 + k, pB + 32 * 128 + k + 16), a2);
    a3 = wmma_f16(af, frag_load(pB + 48 * 128 + k, pB + 48 * 128 + k + 16), a3);
  }

  const int row0 = (lane >> 4) * 8;
#pragma unroll
  for (int nt = 0; nt < 4; ++nt) {
    int col = n0 + nt * 16 + (lane & 15);
    float sc = gamma[col] * rsqrtf(var[col] + BN_EPS);
    float sh = beta[col] - mean[col] * sc;
    v8f acc = (nt == 0) ? a0 : (nt == 1) ? a1 : (nt == 2) ? a2 : a3;
#pragma unroll
    for (int v = 0; v < 8; ++v) {
      size_t r = (size_t)mtile * 16 + row0 + v;
      q[r * 512 + col] = (h16)((acc[v] + qb[col]) * sc + sh);
    }
  }
}

// ---------------------------------------------------------------------------
// scores: for each (head,half) slice hc: [4096,64] x [64,512] -> f32 scores
// s layout: s[(b*8 + hc)*512 + n]   (hc = head*2 + half)
// subkeys [h][c][n][d] is already K-contiguous per column -> perfect B layout.
// ---------------------------------------------------------------------------
__global__ __launch_bounds__(256) void k_scores(const h16* __restrict__ q,
                                                const h16* __restrict__ sk,
                                                float* __restrict__ s) {
  const int lane  = threadIdx.x & 31;
  const int wave  = threadIdx.x >> 5;
  const int mtile = blockIdx.x * 8 + wave;   // 256
  const int n0    = blockIdx.y * 64;         // 8 groups
  const int hc    = blockIdx.z;              // 8 slices
  const int kb    = (lane >> 4) << 3;
  const h16* pA = q  + (size_t)(mtile * 16 + (lane & 15)) * 512 + hc * 64 + kb;
  const h16* pB = sk + ((size_t)hc * 512 + n0 + (lane & 15)) * 64 + kb;

  v8f a0 = {}, a1 = {}, a2 = {}, a3 = {};
#pragma unroll
  for (int k = 0; k < 64; k += 32) {
    v16h af = frag_load(pA + k, pA + k + 16);
    a0 = wmma_f16(af, frag_load(pB + k,           pB + k + 16),           a0);
    a1 = wmma_f16(af, frag_load(pB + 16 * 64 + k, pB + 16 * 64 + k + 16), a1);
    a2 = wmma_f16(af, frag_load(pB + 32 * 64 + k, pB + 32 * 64 + k + 16), a2);
    a3 = wmma_f16(af, frag_load(pB + 48 * 64 + k, pB + 48 * 64 + k + 16), a3);
  }

  const int row0 = (lane >> 4) * 8;
  const int col  = n0 + (lane & 15);
#pragma unroll
  for (int v = 0; v < 8; ++v) {
    size_t r = ((size_t)(mtile * 16 + row0 + v) * 8 + hc) * 512;
    s[r + col]      = a0[v];
    s[r + col + 16] = a1[v];
    s[r + col + 32] = a2[v];
    s[r + col + 48] = a3[v];
  }
}

// ---------------------------------------------------------------------------
// PKM top-k / cartesian top-k / softmax / value gather. One wave per batch row.
// h2[b] = h[b] + sum_heads sum_k softmax(best)_k * values[idx_k]
// ---------------------------------------------------------------------------
#define PKM_WAVES 4
__global__ __launch_bounds__(PKM_WAVES * 32) void k_pkm(
    const float* __restrict__ s, const float* __restrict__ hf,
    const float* __restrict__ values, float* __restrict__ h2) {
  __shared__ float buf[PKM_WAVES][NKEYS];
  __shared__ float topv[PKM_WAVES][2][KNN];
  __shared__ int   topi[PKM_WAVES][2][KNN];
  __shared__ float bw[PKM_WAVES][KNN];
  __shared__ int   bidx[PKM_WAVES][KNN];

  const int lane = threadIdx.x & 31;
  const int wv   = threadIdx.x >> 5;
  const int b    = blockIdx.x * PKM_WAVES + wv;

  float acc[4] = {0.f, 0.f, 0.f, 0.f};

  for (int h = 0; h < HEADS; ++h) {
    // ---- stage 1: top-KNN of each 512-score list ----
    for (int c = 0; c < 2; ++c) {
      const float* sp = s + (((size_t)b * HEADS + h) * 2 + c) * NKEYS;
      for (int i = lane; i < NKEYS; i += 32) buf[wv][i] = sp[i];
      __syncthreads();
      for (int t = 0; t < KNN; ++t) {
        float bv = -1e30f; int bi = 0x7fffffff;
        for (int i = 0; i < NKEYS / 32; ++i) {
          int j = lane * (NKEYS / 32) + i;
          float v = buf[wv][j];
          if (v > bv) { bv = v; bi = j; }
        }
        for (int off = 16; off > 0; off >>= 1) {
          float ov = __shfl_xor(bv, off, 32);
          int   oi = __shfl_xor(bi, off, 32);
          if (ov > bv || (ov == bv && oi < bi)) { bv = ov; bi = oi; }
        }
        if (lane == 0) {
          topv[wv][c][t] = bv; topi[wv][c][t] = bi;
          buf[wv][bi] = -1e30f;
        }
        __syncthreads();
      }
    }

    // ---- stage 2: top-KNN of the 32x32 cartesian sums (lane owns column j) ----
    unsigned used = 0u;
    float s2v = topv[wv][1][lane];
    for (int t = 0; t < KNN; ++t) {
      float bv = -1e30f; int bi = -1;
      for (int i = 0; i < KNN; ++i) {
        if (used & (1u << i)) continue;
        float v = topv[wv][0][i] + s2v;
        if (v > bv) { bv = v; bi = i; }
      }
      int bj = lane;
      for (int off = 16; off > 0; off >>= 1) {
        float ov = __shfl_xor(bv, off, 32);
        int   oi = __shfl_xor(bi, off, 32);
        int   oj = __shfl_xor(bj, off, 32);
        if (ov > bv || (ov == bv && (oj < bj || (oj == bj && oi < bi)))) {
          bv = ov; bi = oi; bj = oj;
        }
      }
      if (lane == bj) used |= (1u << bi);
      if (lane == 0) {
        bw[wv][t]   = bv;
        bidx[wv][t] = topi[wv][0][bi] * NKEYS + topi[wv][1][bj];
      }
      __syncthreads();
    }

    // ---- softmax over the KNN best (lane k owns weight k) ----
    float mx = bw[wv][0];                       // selected in descending order
    float e  = __expf(bw[wv][lane] - mx);
    float ssum = e;
    for (int off = 16; off > 0; off >>= 1) ssum += __shfl_xor(ssum, off, 32);
    bw[wv][lane] = e / ssum;
    __syncthreads();

    // ---- weighted value gather: lane handles dims lane*4 .. lane*4+3 ----
    for (int k = 0; k < KNN; ++k) {
      float w = bw[wv][k];
      const float* vp = values + (size_t)bidx[wv][k] * VDIM + lane * 4;
      acc[0] += w * vp[0]; acc[1] += w * vp[1];
      acc[2] += w * vp[2]; acc[3] += w * vp[3];
    }
    __syncthreads();
  }

  const float* hb = hf + (size_t)b * VDIM + lane * 4;
  float* ob = h2 + (size_t)b * VDIM + lane * 4;
  ob[0] = hb[0] + acc[0]; ob[1] = hb[1] + acc[1];
  ob[2] = hb[2] + acc[2]; ob[3] = hb[3] + acc[3];
}

// ---------------------------------------------------------------------------
// dense2 (128->10) + softmax, one thread per batch row.
// ---------------------------------------------------------------------------
__global__ void k_dense2(const float* __restrict__ h2, const float* __restrict__ w,
                         const float* __restrict__ bias, float* __restrict__ out) {
  int b = blockIdx.x * blockDim.x + threadIdx.x;
  if (b >= BATCH) return;
  const float* hb = h2 + (size_t)b * 128;
  float lg[10];
  float mx = -1e30f;
#pragma unroll
  for (int j = 0; j < 10; ++j) {
    float a = bias[j];
    for (int k = 0; k < 128; ++k) a += hb[k] * w[k * 10 + j];
    lg[j] = a; mx = fmaxf(mx, a);
  }
  float ssum = 0.f;
#pragma unroll
  for (int j = 0; j < 10; ++j) { lg[j] = __expf(lg[j] - mx); ssum += lg[j]; }
  float inv = 1.f / ssum;
#pragma unroll
  for (int j = 0; j < 10; ++j) out[(size_t)b * 10 + j] = lg[j] * inv;
}

// ---------------------------------------------------------------------------
extern "C" void kernel_launch(void* const* d_in, const int* in_sizes, int n_in,
                              void* d_out, int out_size, void* d_ws, size_t ws_size,
                              hipStream_t stream) {
  const float* x      = (const float*)d_in[0];
  const float* w1     = (const float*)d_in[1];
  const float* b1     = (const float*)d_in[2];
  const float* w2     = (const float*)d_in[3];
  const float* b2     = (const float*)d_in[4];
  const float* d1w    = (const float*)d_in[5];
  const float* d1b    = (const float*)d_in[6];
  const float* qw     = (const float*)d_in[7];
  const float* qb     = (const float*)d_in[8];
  const float* gamma  = (const float*)d_in[9];
  const float* beta   = (const float*)d_in[10];
  const float* mean   = (const float*)d_in[11];
  const float* var    = (const float*)d_in[12];
  const float* subk   = (const float*)d_in[13];
  const float* values = (const float*)d_in[14];
  const float* d2w    = (const float*)d_in[15];
  const float* d2b    = (const float*)d_in[16];
  float* out = (float*)d_out;

  char* ws = (char*)d_ws;
  size_t off = 0;
  auto alloc = [&](size_t bytes) -> void* {
    void* p = ws + off;
    off = (off + bytes + 255) & ~(size_t)255;
    return p;
  };

  h16*   h1   = (h16*)  alloc((size_t)BATCH * 26 * 26 * 32 * sizeof(h16));
  h16*   w2t  = (h16*)  alloc((size_t)64 * 288 * sizeof(h16));
  h16*   c2   = (h16*)  alloc((size_t)BATCH * 24 * 24 * 64 * sizeof(h16));
  h16*   a1   = (h16*)  alloc((size_t)BATCH * 9216 * sizeof(h16));
  h16*   d1wt = (h16*)  alloc((size_t)128 * 9216 * sizeof(h16));
  float* hf   = (float*)alloc((size_t)BATCH * 128 * sizeof(float));
  h16*   hh   = (h16*)  alloc((size_t)BATCH * 128 * sizeof(h16));
  h16*   qwt  = (h16*)  alloc((size_t)512 * 128 * sizeof(h16));
  h16*   qf   = (h16*)  alloc((size_t)BATCH * 512 * sizeof(h16));
  h16*   skh  = (h16*)  alloc((size_t)HEADS * 2 * NKEYS * 64 * sizeof(h16));
  float* sbuf = (float*)alloc((size_t)BATCH * 8 * 512 * sizeof(float));
  float* h2   = (float*)alloc((size_t)BATCH * 128 * sizeof(float));
  (void)ws_size; (void)n_in; (void)in_sizes; (void)out_size;

  // conv1 + ReLU -> f16
  {
    int total = BATCH * 26 * 26 * 32;
    k_conv1<<<(total + 255) / 256, 256, 0, stream>>>(x, w1, b1, h1);
  }
  // weight preps (small)
  k_transpose_cast<<<(288 * 64 + 255) / 256, 256, 0, stream>>>(w2, w2t, 288, 64);
  k_transpose_cast<<<(9216 * 128 + 255) / 256, 256, 0, stream>>>(d1w, d1wt, 9216, 128);
  k_transpose_cast<<<(128 * 512 + 255) / 256, 256, 0, stream>>>(qw, qwt, 128, 512);
  k_cast_f16<<<(HEADS * 2 * NKEYS * 64 + 255) / 256, 256, 0, stream>>>(
      subk, skh, HEADS * 2 * NKEYS * 64);

  // conv2 (WMMA implicit GEMM): 147456 m-tiles / 8 waves per block
  k_conv2<<<147456 / 8, 256, 0, stream>>>(h1, w2t, b2, c2);

  // 2x2 maxpool -> flattened dense1 input
  {
    int total = BATCH * 12 * 12 * 64;
    k_pool<<<(total + 255) / 256, 256, 0, stream>>>(c2, a1);
  }

  // dense1 (WMMA + TDM-staged B panel) + bias + ReLU
  k_dense1<<<dim3(32, 2), 256, 0, stream>>>(a1, d1wt, d1b, hf, hh);

  // q projection (WMMA) + batchnorm -> f16
  k_q<<<dim3(32, 8), 256, 0, stream>>>(hh, qwt, qb, gamma, beta, mean, var, qf);

  // sub-key scores (WMMA), 8 (head,half) slices
  k_scores<<<dim3(32, 8, 8), 256, 0, stream>>>(qf, skh, sbuf);

  // PKM: top-k, cartesian top-k, softmax, value gather + residual
  k_pkm<<<BATCH / PKM_WAVES, PKM_WAVES * 32, 0, stream>>>(sbuf, hf, values, h2);

  // dense2 + softmax
  k_dense2<<<BATCH / 256, 256, 0, stream>>>(h2, d2w, d2b, out);
}
